// TransformerAssociator_5927054868528
// MI455X (gfx1250) — matrix operand order repaired and checked by reading, not compiled, not run on hardware
//
#include <hip/hip_runtime.h>

typedef __attribute__((ext_vector_type(16))) _Float16 v16h;
typedef __attribute__((ext_vector_type(8)))  float    v8f;
typedef unsigned int u32x4 __attribute__((ext_vector_type(4)));
typedef int          i32x4 __attribute__((ext_vector_type(4)));
typedef int          i32x8 __attribute__((ext_vector_type(8)));

union AF { v16h v; _Float16 h[16]; };
union CF { v8f  v; float    f[8];  };

__device__ inline v8f wmma16(v16h a, v16h b, v8f c) {
  return __builtin_amdgcn_wmma_f32_16x16x32_f16(false, a, false, b, (short)0, c, false, false);
}

// LDS byte offset of a __shared__ pointer (generic -> addrspace(3) -> int).
__device__ inline unsigned lds_addr_of(const void* p) {
  return (unsigned)(unsigned long long)(__attribute__((address_space(3))) const void*)p;
}

// ---------------------------------------------------------------------------
// TDM: async 2D tile load (64 rows x 32 f16, row stride = strideK halves) from
// global f16 matrix into LDS. LDS rows padded by TDM to 40 halves
// (pad_interval = 16 DWORDs, pad_amount = 4 DWORDs). Tracked by TENSORcnt.
// ---------------------------------------------------------------------------
__device__ inline void tdm_load_b_tile(unsigned lds, const _Float16* gsrc, int strideK) {
  unsigned long long ga = (unsigned long long)gsrc;
  u32x4 g0;
  g0.x = 1u;                                   // count=1, user descriptor
  g0.y = lds;                                  // lds_addr
  g0.z = (unsigned)ga;                         // global_addr[31:0]
  g0.w = (unsigned)(ga >> 32) | 0x80000000u;   // global_addr hi | type=2
  i32x8 g1;
  g1[0] = (1 << 16)    // data_size = 2 bytes
        | (1 << 20)    // pad_enable
        | (3 << 22)    // pad_interval: 16 DWORDs (= one 32-half tile row)
        | (3 << 25);   // pad_amount: 4 DWORDs (8 halves) -> LDS stride 40
  g1[1] = 0;                       // tensor_dim0[15:0] << 16 (dim0 = 1<<20)
  g1[2] = 0x10;                    // tensor_dim0[31:16] | tensor_dim1[15:0]<<16
  g1[3] = 0x10 | (32 << 16);       // tensor_dim1[31:16] | tile_dim0 = 32
  g1[4] = 64;                      // tile_dim1 = 64, tile_dim2 = 0
  g1[5] = strideK;                 // tensor_dim0_stride low 32 (halves)
  g1[6] = 0;
  g1[7] = 0;
  i32x4 gz = {};                   // groups 2/3 unused (2D tensor)
#if __clang_major__ >= 23
  i32x8 g4 = {};
  __builtin_amdgcn_tensor_load_to_lds(g0, g1, gz, gz, g4, 0);
#else
  __builtin_amdgcn_tensor_load_to_lds(g0, g1, gz, gz, 0);
#endif
}

// ---------------------------------------------------------------------------
// GEMM: Y[M,N] = act(alpha * A @ Bt^T + bias + residual)
// A:  [M,K] f32 row-major (converted to f16 while staging).
// Bt: [N,K] f16 row-major (pre-transposed weights) - tiles streamed by TDM,
//     double-buffered, s_wait_tensorcnt-pipelined.
// Block = 256 thr (8 waves), tile 128x64, wave tile 32x32 (2x2 WMMA 16x16x32).
// ---------------------------------------------------------------------------
__global__ __launch_bounds__(256) void gemm_f16w(
    const float* __restrict__ A, const _Float16* __restrict__ Bt,
    const float* __restrict__ bias, const float* __restrict__ R,
    float* __restrict__ Y, int M, int N, int K, int ldY,
    float alpha, int relu)
{
  __shared__ _Float16 As[128 * 40];     // A tile [128][32], stride 40
  __shared__ _Float16 Bs[2][64 * 40];   // B tiles [64][32], stride 40 (TDM pad)
  const int tid  = threadIdx.x;
  const int lane = tid & 31, w = tid >> 5;
  const int wm = w >> 1, wn = w & 1;
  const int l15 = lane & 15, hi = lane >> 4;
  const int m0 = blockIdx.y * 128, n0 = blockIdx.x * 64;

  CF acc[2][2];
#pragma unroll
  for (int i = 0; i < 2; ++i)
#pragma unroll
    for (int j = 0; j < 2; ++j) { v8f z = {}; acc[i][j].v = z; }

  if (w == 0)  // prime the pipeline: tile for k0 = 0 into Bs[0]
    tdm_load_b_tile(lds_addr_of(&Bs[0][0]), Bt + (size_t)n0 * K, K);

  int ib = 0;
  for (int k0 = 0; k0 < K; k0 += 32, ib ^= 1) {
    { // stage A (f32 -> f16)
      const int r = tid >> 1, c = (tid & 1) * 16;
      const float* src = A + (size_t)(m0 + r) * K + k0 + c;
      if (k0 + 32 < K) __builtin_prefetch(src + 32, 0, 3);
      _Float16* dst = &As[r * 40 + c];
#pragma unroll
      for (int i = 0; i < 16; ++i) dst[i] = (_Float16)src[i];
    }
    if (w == 0) {
      if (k0 + 32 < K) {  // issue next tile, then wait for current (in-order)
        tdm_load_b_tile(lds_addr_of(&Bs[ib ^ 1][0]),
                        Bt + (size_t)n0 * K + (k0 + 32), K);
        __builtin_amdgcn_s_wait_tensorcnt(1);
      } else {
        __builtin_amdgcn_s_wait_tensorcnt(0);
      }
    }
    __syncthreads();

    AF a[2], b[2];
#pragma unroll
    for (int t = 0; t < 2; ++t) {
      // A frag (16x32): lane<16 holds K 0-15; lane>=16 holds K 16-31
      const _Float16* pA = &As[(wm * 32 + t * 16 + l15) * 40];
#pragma unroll
      for (int i = 0; i < 8; ++i) {
        a[t].h[i]     = pA[hi * 16 + i];
        a[t].h[8 + i] = pA[hi * 16 + 8 + i];
      }
      // B frag (32x16): frag[t'] = B[k0 + hi*16 + t'][col]
      const _Float16* pB = &Bs[ib][(wn * 32 + t * 16 + l15) * 40 + hi * 16];
#pragma unroll
      for (int i = 0; i < 16; ++i) b[t].h[i] = pB[i];
    }
#pragma unroll
    for (int i = 0; i < 2; ++i)
#pragma unroll
      for (int j = 0; j < 2; ++j)
        acc[i][j].v = wmma16(a[i].v, b[j].v, acc[i][j].v);
    __syncthreads();
  }

#pragma unroll
  for (int i = 0; i < 2; ++i)
#pragma unroll
    for (int j = 0; j < 2; ++j) {
      const int col = n0 + wn * 32 + j * 16 + l15;
      const float bv = bias ? bias[col] : 0.0f;
#pragma unroll
      for (int r = 0; r < 8; ++r) {
        const int row = m0 + wm * 32 + i * 16 + hi * 8 + r;
        float y = acc[i][j].f[r] * alpha + bv;
        if (R)    y += R[(size_t)row * N + col];
        if (relu) y = fmaxf(y, 0.0f);
        Y[(size_t)row * ldY + col] = y;
      }
    }
}

// ---------------------------------------------------------------------------
// Weight prep: W[K,N] f32 -> Wt[N,K] f16 (32x32 LDS tile transpose).
// ---------------------------------------------------------------------------
__global__ __launch_bounds__(256) void transpose_to_f16(
    const float* __restrict__ W, _Float16* __restrict__ Wt, int K, int N)
{
  __shared__ float t[32][33];
  const int k0 = blockIdx.x * 32, n0 = blockIdx.y * 32;
  const int tx = threadIdx.x & 31, ty = threadIdx.x >> 5;
  for (int i = ty; i < 32; i += 8)
    t[i][tx] = W[(size_t)(k0 + i) * N + n0 + tx];
  __syncthreads();
  for (int i = ty; i < 32; i += 8)
    Wt[(size_t)(n0 + i) * K + k0 + tx] = (_Float16)t[tx][i];
}

__global__ __launch_bounds__(256) void f32_to_f16_copy(
    const float* __restrict__ s, _Float16* __restrict__ d)
{
  const size_t i = (size_t)blockIdx.x * 256 + threadIdx.x;
  d[i] = (_Float16)s[i];
}

// ---------------------------------------------------------------------------
// Attention: one wave per (seq, head, 16-query tile). Full 16x1024 score block
// kept in LDS (no HBM traffic for S/P). S = Q K^T / sqrt(32); O = softmax(S) V.
// ---------------------------------------------------------------------------
#define ATT_SMEM (16*1024*4 + 16*1024*2 + 32*40*2 + 16*4)

__global__ __launch_bounds__(32) void attn_flash(
    const float* __restrict__ Qb, const float* __restrict__ Kb,
    const float* __restrict__ Vb, float* __restrict__ Ob, int cross)
{
  extern __shared__ char smem[];
  float*    S    = (float*)smem;                            // [16][1024]
  _Float16* P    = (_Float16*)(smem + 65536);               // [16][1024]
  _Float16* Vt   = (_Float16*)(smem + 65536 + 32768);       // [32 d][40]
  float*    invL = (float*)(smem + 65536 + 32768 + 2560);   // [16]

  const int qt = blockIdx.x, h = blockIdx.y, seq = blockIdx.z;
  const int kv = cross ? (seq ^ 1) : seq;
  const int lane = threadIdx.x, l15 = lane & 15, hi = lane >> 4;

  const float* Qp = Qb + ((size_t)seq * 1024 + qt * 16) * 256 + h * 32;
  const float* Kp = Kb + (size_t)kv * 1024 * 256 + h * 32;
  const float* Vp = Vb + (size_t)kv * 1024 * 256 + h * 32;

  AF qa; // Q fragment 16x32 (K = head dim)
#pragma unroll
  for (int i = 0; i < 8; ++i) {
    qa.h[i]     = (_Float16)Qp[(size_t)l15 * 256 + hi * 16 + i];
    qa.h[8 + i] = (_Float16)Qp[(size_t)l15 * 256 + hi * 16 + 8 + i];
  }
  const float scale = 0.1767766952966369f; // 1/sqrt(32)

  for (int kt = 0; kt < 64; ++kt) {        // 16 keys per WMMA
    AF kb; // B frag: frag[t] = K[key = kt*16+l15][d = hi*16+t]
    const float* krow = Kp + (size_t)(kt * 16 + l15) * 256 + hi * 16;
#pragma unroll
    for (int t = 0; t < 16; ++t) kb.h[t] = (_Float16)krow[t];
    CF s; { v8f z = {}; s.v = z; }
    s.v = wmma16(qa.v, kb.v, s.v);
#pragma unroll
    for (int r = 0; r < 8; ++r)
      S[(hi * 8 + r) * 1024 + kt * 16 + l15] = s.f[r] * scale;
  }
  __syncthreads();

  // softmax: lane pair (l15, l15+16) splits the 1024 columns of row l15
  const int row = l15;
  float m = -__builtin_inff();
  for (int j = hi * 512; j < hi * 512 + 512; ++j) m = fmaxf(m, S[row * 1024 + j]);
  m = fmaxf(m, __shfl_xor(m, 16));
  float ss = 0.0f;
  for (int j = hi * 512; j < hi * 512 + 512; ++j) {
    float e = __expf(S[row * 1024 + j] - m);
    P[row * 1024 + j] = (_Float16)e;
    ss += e;
  }
  ss += __shfl_xor(ss, 16);
  if (hi == 0) invL[row] = 1.0f / ss;
  __syncthreads();

  CF o0, o1; { v8f z = {}; o0.v = z; o1.v = z; }
  for (int kt = 0; kt < 32; ++kt) {        // 32 keys per step
    const float* vrow = Vp + (size_t)(kt * 32 + lane) * 256; // lane = key
#pragma unroll
    for (int d = 0; d < 32; ++d) Vt[d * 40 + lane] = (_Float16)vrow[d];
    __syncthreads();
    AF pa; // P fragment 16x32
    const _Float16* pr = &P[row * 1024 + kt * 32];
#pragma unroll
    for (int i = 0; i < 8; ++i) {
      pa.h[i]     = pr[hi * 16 + i];
      pa.h[8 + i] = pr[hi * 16 + 8 + i];
    }
    AF b0, b1; // V frags: frag[t] = V[key=hi*16+t][d]
#pragma unroll
    for (int t = 0; t < 16; ++t) {
      b0.h[t] = Vt[l15 * 40 + hi * 16 + t];
      b1.h[t] = Vt[(16 + l15) * 40 + hi * 16 + t];
    }
    o0.v = wmma16(pa.v, b0.v, o0.v);
    o1.v = wmma16(pa.v, b1.v, o1.v);
    __syncthreads();
  }

  float* Op = Ob + ((size_t)seq * 1024 + qt * 16) * 256 + h * 32;
#pragma unroll
  for (int r = 0; r < 8; ++r) {
    const int rr = hi * 8 + r;
    const float inv = invL[rr];
    Op[(size_t)rr * 256 + l15]      = o0.f[r] * inv;
    Op[(size_t)rr * 256 + 16 + l15] = o1.f[r] * inv;
  }
}

// ---------------------------------------------------------------------------
// LayerNorm over dim 256: one wave per row, 8 rows per block.
// ---------------------------------------------------------------------------
__global__ __launch_bounds__(256) void layernorm256(
    const float* __restrict__ Xin, const float* __restrict__ g,
    const float* __restrict__ b, float* __restrict__ Y)
{
  const int w = threadIdx.x >> 5, lane = threadIdx.x & 31;
  const int row = blockIdx.x * 8 + w;
  const float* x = Xin + (size_t)row * 256;
  float vals[8], sum = 0.0f;
#pragma unroll
  for (int i = 0; i < 8; ++i) { vals[i] = x[lane + 32 * i]; sum += vals[i]; }
#pragma unroll
  for (int o = 16; o; o >>= 1) sum += __shfl_xor(sum, o);
  const float mean = sum * (1.0f / 256.0f);
  float var = 0.0f;
#pragma unroll
  for (int i = 0; i < 8; ++i) { float d = vals[i] - mean; var += d * d; }
#pragma unroll
  for (int o = 16; o; o >>= 1) var += __shfl_xor(var, o);
  const float inv = rsqrtf(var * (1.0f / 256.0f) + 1e-5f);
  float* y = Y + (size_t)row * 256;
#pragma unroll
  for (int i = 0; i < 8; ++i) {
    int c = lane + 32 * i;
    y[c] = g[c] * (vals[i] - mean) * inv + b[c];
  }
}

// ---------------------------------------------------------------------------
// CNN encoder: one block per keypoint patch; all intermediates in LDS.
// token = img*2048 + stream*1024 + point  ->  seq = img*2 + stream.
// ---------------------------------------------------------------------------
__global__ __launch_bounds__(256) void conv_encoder(
    const float* __restrict__ bgrs0, const float* __restrict__ bgrs1,
    const float* __restrict__ pe0, const float* __restrict__ pe1,
    const float* __restrict__ W1, const float* __restrict__ B1,
    const float* __restrict__ W2, const float* __restrict__ B2,
    const float* __restrict__ W3, const float* __restrict__ B3,
    const float* __restrict__ W4, const float* __restrict__ B4,
    float* __restrict__ feat)
{
  __shared__ float sin_[768];   // 3x16x16
  __shared__ float c1[2048];    // 32x8x8
  __shared__ float c2[1024];    // 64x4x4
  __shared__ float c3[512];     // 128x2x2
  const int tid = threadIdx.x, blk = blockIdx.x;
  const int img = blk >> 11, str = (blk >> 10) & 1, pt = blk & 1023;
  const float* in = (str ? bgrs1 : bgrs0) + ((size_t)img * 1024 + pt) * 768;
  for (int i = tid; i < 768; i += 256) sin_[i] = in[i];
  __syncthreads();

  for (int o = tid; o < 2048; o += 256) {          // conv1: 3->32, 16->8
    int oc = o >> 6, oy = (o >> 3) & 7, ox = o & 7;
    float a = B1[oc];
    for (int ic = 0; ic < 3; ++ic)
      for (int ky = 0; ky < 3; ++ky) {
        int iy = oy * 2 - 1 + ky; if ((unsigned)iy >= 16u) continue;
        for (int kx = 0; kx < 3; ++kx) {
          int ix = ox * 2 - 1 + kx; if ((unsigned)ix >= 16u) continue;
          a += sin_[ic * 256 + iy * 16 + ix] * W1[((oc * 3 + ic) * 3 + ky) * 3 + kx];
        }
      }
    c1[o] = fmaxf(a, 0.0f);
  }
  __syncthreads();

  for (int o = tid; o < 1024; o += 256) {          // conv2: 32->64, 8->4
    int oc = o >> 4, oy = (o >> 2) & 3, ox = o & 3;
    float a = B2[oc];
    for (int ic = 0; ic < 32; ++ic)
      for (int ky = 0; ky < 3; ++ky) {
        int iy = oy * 2 - 1 + ky; if ((unsigned)iy >= 8u) continue;
        for (int kx = 0; kx < 3; ++kx) {
          int ix = ox * 2 - 1 + kx; if ((unsigned)ix >= 8u) continue;
          a += c1[ic * 64 + iy * 8 + ix] * W2[((oc * 32 + ic) * 3 + ky) * 3 + kx];
        }
      }
    c2[o] = fmaxf(a, 0.0f);
  }
  __syncthreads();

  for (int o = tid; o < 512; o += 256) {           // conv3: 64->128, 4->2
    int oc = o >> 2, oy = (o >> 1) & 1, ox = o & 1;
    float a = B3[oc];
    for (int ic = 0; ic < 64; ++ic)
      for (int ky = 0; ky < 3; ++ky) {
        int iy = oy * 2 - 1 + ky; if ((unsigned)iy >= 4u) continue;
        for (int kx = 0; kx < 3; ++kx) {
          int ix = ox * 2 - 1 + kx; if ((unsigned)ix >= 4u) continue;
          a += c2[ic * 16 + iy * 4 + ix] * W3[((oc * 64 + ic) * 3 + ky) * 3 + kx];
        }
      }
    c3[o] = fmaxf(a, 0.0f);
  }
  __syncthreads();

  {                                                // conv4: 128->256, 2->1 (no relu)
    int oc = tid;
    float a = B4[oc];
    for (int ic = 0; ic < 128; ++ic)
      for (int ky = 1; ky < 3; ++ky)
        for (int kx = 1; kx < 3; ++kx)
          a += c3[ic * 4 + (ky - 1) * 2 + (kx - 1)] * W4[((oc * 128 + ic) * 3 + ky) * 3 + kx];
    const float* pe = (str ? pe1 : pe0) + ((size_t)img * 1024 + pt) * 256;
    feat[(size_t)blk * 256 + oc] = a + pe[oc];
  }
}

// ---------------------------------------------------------------------------
// Sinkhorn (log-domain optimal transport), m = n = 1024.
// ---------------------------------------------------------------------------
#define NORM_CONST (-7.6246189861593985f)  // -log(2048)
#define LOG1024    (6.9314718055994531f)   //  log(1024)

__global__ __launch_bounds__(256) void sink_init(
    float* __restrict__ Z, float* __restrict__ u, float* __restrict__ v,
    const float* __restrict__ bin)
{
  const int img = blockIdx.y;
  const int i = blockIdx.x * 256 + threadIdx.x;
  if (i > 1024) return;
  float* Zi = Z + (size_t)img * 1025 * 1025;
  const float a = bin[0];
  Zi[(size_t)1024 * 1025 + i] = a;   // last row
  Zi[(size_t)i * 1025 + 1024] = a;   // last col
  u[img * 1025 + i] = 0.0f;
  v[img * 1025 + i] = 0.0f;
}

__global__ __launch_bounds__(256) void sink_row(
    const float* __restrict__ Z, const float* __restrict__ v, float* __restrict__ u)
{
  const int img = blockIdx.y, i = blockIdx.x, tid = threadIdx.x;
  const float* Zr = Z + (size_t)img * 1025 * 1025 + (size_t)i * 1025;
  const float* vv = v + img * 1025;
  float m = -__builtin_inff(), s = 0.0f;
  for (int j = tid; j < 1025; j += 256) {
    float x = Zr[j] + vv[j];
    if (x > m) { s = s * __expf(m - x) + 1.0f; m = x; }
    else       { s += __expf(x - m); }
  }
#pragma unroll
  for (int o = 16; o; o >>= 1) {
    float m2 = __shfl_xor(m, o), s2 = __shfl_xor(s, o);
    float M = fmaxf(m, m2);
    s = s * __expf(m - M) + s2 * __expf(m2 - M);
    m = M;
  }
  __shared__ float sm[8], ss[8];
  if ((tid & 31) == 0) { sm[tid >> 5] = m; ss[tid >> 5] = s; }
  __syncthreads();
  if (tid == 0) {
    float M = sm[0], Sx = ss[0];
    for (int w2 = 1; w2 < 8; ++w2) {
      float M2 = fmaxf(M, sm[w2]);
      Sx = Sx * __expf(M - M2) + ss[w2] * __expf(sm[w2] - M2);
      M = M2;
    }
    float lse = M + __logf(Sx);
    float mu = (i < 1024) ? NORM_CONST : (LOG1024 + NORM_CONST);
    u[img * 1025 + i] = mu - lse;
  }
}

__global__ __launch_bounds__(256) void sink_col(
    const float* __restrict__ Z, const float* __restrict__ u, float* __restrict__ v)
{
  const int img = blockIdx.y;
  const int j = blockIdx.x * 256 + threadIdx.x;
  if (j > 1024) return;
  const float* Zi = Z + (size_t)img * 1025 * 1025;
  const float* uu = u + img * 1025;
  float m = -__builtin_inff(), s = 0.0f;
  for (int i = 0; i < 1025; ++i) {         // coalesced across threads
    float x = Zi[(size_t)i * 1025 + j] + uu[i];
    if (x > m) { s = s * __expf(m - x) + 1.0f; m = x; }
    else       { s += __expf(x - m); }
  }
  float lse = m + __logf(s);
  float nu = (j < 1024) ? NORM_CONST : (LOG1024 + NORM_CONST);
  v[img * 1025 + j] = nu - lse;
}

__global__ __launch_bounds__(256) void sink_final(
    const float* __restrict__ Z, const float* __restrict__ u,
    const float* __restrict__ v, float* __restrict__ out)
{
  const int img = blockIdx.y;
  const size_t idx = (size_t)blockIdx.x * 256 + threadIdx.x;
  if (idx >= (size_t)1025 * 1025) return;
  const int i = (int)(idx / 1025), j = (int)(idx % 1025);
  out[(size_t)img * 1025 * 1025 + idx] =
      Z[(size_t)img * 1025 * 1025 + idx] + u[img * 1025 + i] + v[img * 1025 + j] - NORM_CONST;
}

// ---------------------------------------------------------------------------
// Host orchestration. Input flattening (jax pytree order):
//   0 bgrs_0, 1 pe_0, 2 is_mask_0, 3 bgrs_1, 4 pe_1, 5 is_mask_1,
//   6..13 conv (W1,b1,W2,b2,W3,b3,W4,b4),
//   14 + 16*L + k with sorted dict keys:
//     0 bk 1 bo 2 bq 3 bv 4 ff_b1 5 ff_b2 6 ff_w1 7 ff_w2
//     8 ln1_b 9 ln1_g 10 ln2_b 11 ln2_g 12 wk 13 wo 14 wq 15 wv
//   110 bin_score
// ---------------------------------------------------------------------------
extern "C" void kernel_launch(void* const* d_in, const int* in_sizes, int n_in,
                              void* d_out, int out_size, void* d_ws, size_t ws_size,
                              hipStream_t stream) {
  (void)in_sizes; (void)n_in; (void)out_size; (void)ws_size;
  const float* bgrs0 = (const float*)d_in[0];
  const float* pe0   = (const float*)d_in[1];
  const float* bgrs1 = (const float*)d_in[3];
  const float* pe1   = (const float*)d_in[4];
  const float* cw[4]; const float* cb[4];
  for (int i = 0; i < 4; ++i) {
    cw[i] = (const float*)d_in[6 + 2 * i];
    cb[i] = (const float*)d_in[7 + 2 * i];
  }
  auto lp = [&](int L, int k) { return (const float*)d_in[14 + 16 * L + k]; };
  const float* bin = (const float*)d_in[110];

  float* p = (float*)d_ws;
  float* X  = p; p += (size_t)4096 * 256;
  float* Qb = p; p += (size_t)4096 * 256;
  float* Kb = p; p += (size_t)4096 * 256;
  float* Vb = p; p += (size_t)4096 * 256;
  float* Ob = p; p += (size_t)4096 * 256;
  float* T  = p; p += (size_t)4096 * 256;
  float* X1 = p; p += (size_t)4096 * 256;
  float* H  = p; p += (size_t)4096 * 1024;
  float* Z  = p; p += (size_t)2 * 1025 * 1025;
  float* uu = p; p += (size_t)2 * 1025;
  float* vv = p; p += (size_t)2 * 1026;   // keep next region 16B-aligned
  _Float16* hbase = (_Float16*)p;
  // per-layer f16 transposed weights: wq wk wv wo (65536 ea), fw1/fw2 (262144 ea)
  const size_t LSTRIDE = 786432;
  _Float16* Wt = hbase;                         // 6 * LSTRIDE halves
  _Float16* Xh = hbase + 6 * LSTRIDE;           // 4096*256 halves

  // ---- weight prep (once per launch; ~9.4 MB f16, L2-resident afterwards)
  for (int L = 0; L < 6; ++L) {
    _Float16* wb = Wt + (size_t)L * LSTRIDE;
    transpose_to_f16<<<dim3(8, 8),  256, 0, stream>>>(lp(L,14), wb,            256, 256);  // wq
    transpose_to_f16<<<dim3(8, 8),  256, 0, stream>>>(lp(L,12), wb + 65536,    256, 256);  // wk
    transpose_to_f16<<<dim3(8, 8),  256, 0, stream>>>(lp(L,15), wb + 131072,   256, 256);  // wv
    transpose_to_f16<<<dim3(8, 8),  256, 0, stream>>>(lp(L,13), wb + 196608,   256, 256);  // wo
    transpose_to_f16<<<dim3(8, 32), 256, 0, stream>>>(lp(L,6),  wb + 262144,   256, 1024); // ff_w1
    transpose_to_f16<<<dim3(32, 8), 256, 0, stream>>>(lp(L,7),  wb + 524288,  1024, 256);  // ff_w2
  }

  conv_encoder<<<4096, 256, 0, stream>>>(bgrs0, bgrs1, pe0, pe1,
      cw[0], cb[0], cw[1], cb[1], cw[2], cb[2], cw[3], cb[3], X);

  const dim3 gp(256 / 64, 4096 / 128);          // N=256 GEMMs, M=4096
  const dim3 gf(1024 / 64, 4096 / 128);         // N=1024 FFN GEMM

  for (int L = 0; L < 6; ++L) {
    const int cross = L & 1;
    _Float16* wb = Wt + (size_t)L * LSTRIDE;
    gemm_f16w<<<gp, 256, 0, stream>>>(X, wb,          lp(L,2), nullptr, Qb,
                                      4096, 256, 256, 256, 1.0f, 0);
    gemm_f16w<<<gp, 256, 0, stream>>>(X, wb + 65536,  lp(L,0), nullptr, Kb,
                                      4096, 256, 256, 256, 1.0f, 0);
    gemm_f16w<<<gp, 256, 0, stream>>>(X, wb + 131072, lp(L,3), nullptr, Vb,
                                      4096, 256, 256, 256, 1.0f, 0);
    attn_flash<<<dim3(64, 8, 4), 32, ATT_SMEM, stream>>>(Qb, Kb, Vb, Ob, cross);
    gemm_f16w<<<gp, 256, 0, stream>>>(Ob, wb + 196608, lp(L,1), X, T,
                                      4096, 256, 256, 256, 1.0f, 0);
    layernorm256<<<512, 256, 0, stream>>>(T, lp(L,9), lp(L,8), X1);
    gemm_f16w<<<gf, 256, 0, stream>>>(X1, wb + 262144, lp(L,4), nullptr, H,
                                      4096, 1024, 256, 1024, 1.0f, 1);
    gemm_f16w<<<gp, 256, 0, stream>>>(H, wb + 524288, lp(L,5), X1, T,
                                      4096, 256, 1024, 256, 1.0f, 0);
    layernorm256<<<512, 256, 0, stream>>>(T, lp(L,11), lp(L,10), X);
  }

  // d1 features to f16 (already [N,K] layout for the transposed-B GEMM)
  f32_to_f16_copy<<<4096, 256, 0, stream>>>(X, Xh);

  // scores: Z interior = (d0 @ d1^T) / sqrt(256), written with pitch 1025
  for (int img = 0; img < 2; ++img) {
    const float*    A  = X  + (size_t)(img * 2 + 0) * 1024 * 256;
    const _Float16* Bm = Xh + (size_t)(img * 2 + 1) * 1024 * 256;
    gemm_f16w<<<dim3(1024 / 64, 1024 / 128), 256, 0, stream>>>(
        A, Bm, nullptr, nullptr, Z + (size_t)img * 1025 * 1025,
        1024, 1024, 256, 1025, 0.0625f, 0);
  }

  sink_init<<<dim3(5, 2), 256, 0, stream>>>(Z, uu, vv, bin);
  for (int it = 0; it < 20; ++it) {
    sink_row<<<dim3(1025, 2), 256, 0, stream>>>(Z, vv, uu);
    sink_col<<<dim3(5, 2), 256, 0, stream>>>(Z, uu, vv);
  }
  sink_final<<<dim3(4105, 2), 256, 0, stream>>>(Z, uu, vv, (float*)d_out);
}
